// Decoder_84275848282402
// MI455X (gfx1250) — matrix operand order repaired and checked by reading, not compiled, not run on hardware
//
#include <hip/hip_runtime.h>
#include <hip/hip_bf16.h>
#include <math.h>

// ---------------------------------------------------------------------------
// Decoder step for B=64, T=512, H=1024, E=512, V=32000 on MI455X (gfx1250).
//
// Roofline: ~0.57 GB traffic -> ~24us @ 23.3 TB/s HBM. The dominant GEMM
// (enc @ We^T inside the attention energy, 68.7 GFLOP) runs as bf16 WMMA
// (v_wmma_f32_16x16x32_bf16, f32 accumulate); f32->bf16 conversion is done
// in-register via native converts (no extra memory pass). The [B,T,H] energy
// tensor (134 MB) is never materialized: relu(.+hWh)·v fused in the epilogue.
// ---------------------------------------------------------------------------

#define Bc 64
#define Tc 512
#define Hc 1024
#define Ec 512
#define Vc 32000
#define BTc (Bc * Tc)

typedef __attribute__((ext_vector_type(16))) __bf16 v16bf;
typedef __attribute__((ext_vector_type(16))) float  v16f;
typedef __attribute__((ext_vector_type(4)))  __bf16 v4bf;
typedef __attribute__((ext_vector_type(4)))  float  v4f;
typedef __attribute__((ext_vector_type(8)))  float  v8f;

union BF16x16 { v16bf v; unsigned int u[8]; };
union BF16x4  { v4bf  v; uint2 u2; };

// 16 f32 -> 16 bf16 using native packed converts (compiler picks
// v_cvt_pk_bf16_f32-class ops on gfx1250 instead of scalar bit-twiddling).
__device__ __forceinline__ v16bf cvt16(float4 a, float4 b, float4 c, float4 e) {
    v16f f;
    f[0]  = a.x; f[1]  = a.y; f[2]  = a.z; f[3]  = a.w;
    f[4]  = b.x; f[5]  = b.y; f[6]  = b.z; f[7]  = b.w;
    f[8]  = c.x; f[9]  = c.y; f[10] = c.z; f[11] = c.w;
    f[12] = e.x; f[13] = e.y; f[14] = e.z; f[15] = e.w;
    return __builtin_convertvector(f, v16bf);
}

__device__ __forceinline__ uint2 cvt4_pack(float4 s) {
    v4f f; f[0] = s.x; f[1] = s.y; f[2] = s.z; f[3] = s.w;
    BF16x4 p; p.v = __builtin_convertvector(f, v4bf);
    return p.u2;
}

__device__ __forceinline__ v8f wmma_bf16(v16bf a, v16bf b, v8f c) {
    return __builtin_amdgcn_wmma_f32_16x16x32_bf16(
        false, a, false, b, (short)0, c, false, false);
}

// ---------------------------------------------------------------------------
// Generic 1-wave WMMA GEMM: C[M,N] = A[M,K] @ W[N,K]^T + bias.
// grid = (N/16, M/16), block = 32 (one wave32 -> one 16x16 tile).
// A/W are f32 row-major; converted to bf16 in-register.
// Fragment layouts per CDNA5 ISA 7.12.2 (wave32).
// ---------------------------------------------------------------------------
__global__ void gemm16_wmma(const float* __restrict__ A, int lda,
                            const float* __restrict__ W, int ldw,
                            const float* __restrict__ bias,
                            float* __restrict__ C, int ldc, int K) {
    const int lane = threadIdx.x;
    const int lo = lane & 15;       // N for B/C, M for A
    const int hi = lane >> 4;       // lane-group select
    const int nt = blockIdx.x, mt = blockIdx.y;

    const float* Arow = A + (size_t)(mt * 16 + lo) * lda;
    const float* Wrow = W + (size_t)(nt * 16 + lo) * ldw;

    v8f c;
    for (int j = 0; j < 8; ++j) c[j] = 0.0f;

    for (int k0 = 0; k0 < K; k0 += 32) {
        __builtin_prefetch(Wrow + k0 + 32, 0, 3);
        // A 16x32 bf16: lanes<16 hold K k0+[0,8)&k0+[16,24); lanes>=16: +8
        const float4* pa0 = (const float4*)(Arow + k0 + hi * 8);
        const float4* pa1 = (const float4*)(Arow + k0 + 16 + hi * 8);
        const v16bf af = cvt16(pa0[0], pa0[1], pa1[0], pa1[1]);
        // B 32x16 bf16: same K order as A (k0+[0,8)&k0+[16,24); lanes>=16: +8)
        const float4* pb0 = (const float4*)(Wrow + k0 + hi * 8);
        const float4* pb1 = (const float4*)(Wrow + k0 + 16 + hi * 8);
        const v16bf bf = cvt16(pb0[0], pb0[1], pb1[0], pb1[1]);
        c = wmma_bf16(af, bf, c);
    }

    const int n = nt * 16 + lo;
    const float bi = bias[n];
    float* Cout = C + (size_t)(mt * 16 + hi * 8) * ldc + n;
    for (int j = 0; j < 8; ++j) {
        Cout[(size_t)j * ldc] = c[j] + bi;   // C layout: VGPR j <-> M
    }
}

// ---------------------------------------------------------------------------
// Fused attention scores: scores[r] = sum_n relu( (enc @ We^T)[r,n]
//                                              + hWh[b,n] ) * v[n]
// r = b*T + t. One WG = 4 waves, 64 rows of enc staged as bf16 in 128 KB LDS
// (CDNA5 WGP has 320 KB). Waves split the 64 N-tiles (16 each); loop order
// N -> K -> M-subtile so each We K-slice is loaded once per N-tile.
// ---------------------------------------------------------------------------
__global__ void attn_scores_wmma(const float* __restrict__ enc,   // [T,B,H]
                                 const float* __restrict__ attnW, // [H,2H]
                                 const float* __restrict__ hWh,   // [B,H]
                                 const float* __restrict__ vvec,  // [H]
                                 float* __restrict__ scores) {    // [B*T]
    extern __shared__ unsigned short Asb[];  // 64 x 1024 bf16 = 128 KB
    __shared__ float score_sm[64];

    const int tid = threadIdx.x;             // 0..127
    const int wg  = blockIdx.x;              // 0..511
    const int b   = wg >> 3;                 // 64 rows never cross a batch row

    // Stage 64 enc rows (f32 -> bf16), coalesced float4 loads.
    for (int idx = tid; idx < (64 * Hc) / 4; idx += 128) {
        const int rl = idx >> 8;             // row 0..63 (256 float4 per row)
        const int c4 = idx & 255;
        const int t  = (wg * 64 + rl) & (Tc - 1);
        const float4 s =
            ((const float4*)(enc + ((size_t)t * Bc + b) * Hc))[c4];
        ((uint2*)Asb)[idx] = cvt4_pack(s);
    }
    if (tid < 64) score_sm[tid] = 0.0f;
    __syncthreads();

    const int wv = tid >> 5, lane = tid & 31;
    const int lo = lane & 15, hi = lane >> 4;

    float part[4][8];
    for (int ms = 0; ms < 4; ++ms)
        for (int j = 0; j < 8; ++j) part[ms][j] = 0.0f;

    for (int ntl = 0; ntl < 16; ++ntl) {
        const int nt = wv * 16 + ntl;
        const int n  = nt * 16 + lo;
        const float* Wrow = attnW + (size_t)n * (2 * Hc) + Hc;  // We part
        const float vn  = vvec[n];
        const float hwn = hWh[b * Hc + n];

        v8f c[4];
        for (int ms = 0; ms < 4; ++ms)
            for (int j = 0; j < 8; ++j) c[ms][j] = 0.0f;

        for (int k0 = 0; k0 < Hc; k0 += 32) {
            __builtin_prefetch(Wrow + k0 + 32, 0, 3);
            const float4* pb0 = (const float4*)(Wrow + k0 + hi * 8);
            const float4* pb1 = (const float4*)(Wrow + k0 + 16 + hi * 8);
            const v16bf bf = cvt16(pb0[0], pb0[1], pb1[0], pb1[1]);
            for (int ms = 0; ms < 4; ++ms) {
                const int m = ms * 16 + lo;
                const uint4* pa0 =
                    (const uint4*)(Asb + m * Hc + k0 + hi * 8);
                const uint4* pa1 =
                    (const uint4*)(Asb + m * Hc + k0 + 16 + hi * 8);
                BF16x16 af;
                uint4 q0 = pa0[0], q1 = pa1[0];
                af.u[0] = q0.x; af.u[1] = q0.y; af.u[2] = q0.z; af.u[3] = q0.w;
                af.u[4] = q1.x; af.u[5] = q1.y; af.u[6] = q1.z; af.u[7] = q1.w;
                c[ms] = wmma_bf16(af.v, bf, c[ms]);
            }
        }
        // Fused epilogue: relu(energy + hWh) * v, accumulated per row.
        for (int ms = 0; ms < 4; ++ms)
            for (int j = 0; j < 8; ++j) {
                float e = c[ms][j] + hwn;
                e = e > 0.0f ? e : 0.0f;
                part[ms][j] += e * vn;
            }
    }

    // Reduce over the 16 N-lanes of each lane group, then across waves in LDS.
    for (int ms = 0; ms < 4; ++ms)
        for (int j = 0; j < 8; ++j) {
            float s = part[ms][j];
            s += __shfl_xor(s, 1, 32);
            s += __shfl_xor(s, 2, 32);
            s += __shfl_xor(s, 4, 32);
            s += __shfl_xor(s, 8, 32);
            if (lo == 0) atomicAdd(&score_sm[ms * 16 + hi * 8 + j], s);
        }
    __syncthreads();
    if (tid < 64) scores[wg * 64 + tid] = score_sm[tid];
}

// softmax over T per batch row -> attention weights (written to output tensor)
__global__ void softmax_T(const float* __restrict__ scores,
                          float* __restrict__ attnw) {
    const int b = blockIdx.x, tid = threadIdx.x;
    __shared__ float red[256];
    float s0 = scores[b * Tc + tid];
    float s1 = scores[b * Tc + 256 + tid];
    red[tid] = fmaxf(s0, s1);
    __syncthreads();
    for (int o = 128; o; o >>= 1) {
        if (tid < o) red[tid] = fmaxf(red[tid], red[tid + o]);
        __syncthreads();
    }
    const float m = red[0];
    __syncthreads();
    float e0 = __expf(s0 - m), e1 = __expf(s1 - m);
    red[tid] = e0 + e1;
    __syncthreads();
    for (int o = 128; o; o >>= 1) {
        if (tid < o) red[tid] += red[tid + o];
        __syncthreads();
    }
    const float inv = 1.0f / red[0];
    attnw[b * Tc + tid]       = e0 * inv;
    attnw[b * Tc + 256 + tid] = e1 * inv;
}

// context[b,h] = sum_t w[b,t] * enc[t,b,h]; coalesced float4 stream of enc
__global__ void context_k(const float* __restrict__ enc,
                          const float* __restrict__ attnw,
                          float* __restrict__ ctx) {
    const int b = blockIdx.x, tid = threadIdx.x;
    float ax = 0, ay = 0, az = 0, aw = 0;
    for (int t = 0; t < Tc; ++t) {
        const float w = attnw[b * Tc + t];
        const float4 e = ((const float4*)(enc + ((size_t)t * Bc + b) * Hc))[tid];
        ax += w * e.x; ay += w * e.y; az += w * e.z; aw += w * e.w;
    }
    ((float4*)(ctx + (size_t)b * Hc))[tid] = make_float4(ax, ay, az, aw);
}

// rnn_in = concat(embed_W[x[b]], context)  ([B, E+H])
__global__ void build_rnnin(const int* __restrict__ x,
                            const float* __restrict__ embW,
                            const float* __restrict__ ctx,
                            float* __restrict__ rnnin) {
    const int idx = blockIdx.x * 256 + threadIdx.x;   // B*(E+H)
    const int b = idx / (Ec + Hc), e = idx % (Ec + Hc);
    float val = (e < Ec) ? embW[(size_t)x[b] * Ec + e]
                         : ctx[(size_t)b * Hc + (e - Ec)];
    rnnin[idx] = val;
}

__device__ __forceinline__ float sigm(float x) {
    return 1.0f / (1.0f + __expf(-x));
}

// GRU gate fusion (PyTorch r,z,n order); also builds hc = concat(h_new, ctx)
__global__ void gru_gates(const float* __restrict__ gi,
                          const float* __restrict__ gh,
                          const float* __restrict__ hprev,
                          const float* __restrict__ ctx,
                          float* __restrict__ h_out,
                          float* __restrict__ hc) {
    const int idx = blockIdx.x * 256 + threadIdx.x;   // B*H
    const int b = idx >> 10, hh = idx & (Hc - 1);
    const size_t g = (size_t)b * 3 * Hc + hh;
    const float r = sigm(gi[g] + gh[g]);
    const float z = sigm(gi[g + Hc] + gh[g + Hc]);
    const float n = tanhf(gi[g + 2 * Hc] + r * gh[g + 2 * Hc]);
    const float hp = hprev[idx];
    const float hn = (1.0f - z) * n + z * hp;
    h_out[idx] = hn;
    hc[(size_t)b * 2 * Hc + hh]      = hn;
    hc[(size_t)b * 2 * Hc + Hc + hh] = ctx[idx];
}

// row-wise log_softmax over V=32000
__global__ void logsoftmax_V(const float* __restrict__ logits,
                             float* __restrict__ out) {
    const int b = blockIdx.x, tid = threadIdx.x;
    __shared__ float red[256];
    const float* row = logits + (size_t)b * Vc;
    float m = -3.4e38f;
    for (int i = tid; i < Vc; i += 256) m = fmaxf(m, row[i]);
    red[tid] = m;
    __syncthreads();
    for (int o = 128; o; o >>= 1) {
        if (tid < o) red[tid] = fmaxf(red[tid], red[tid + o]);
        __syncthreads();
    }
    m = red[0];
    __syncthreads();
    float s = 0.0f;
    for (int i = tid; i < Vc; i += 256) s += __expf(row[i] - m);
    red[tid] = s;
    __syncthreads();
    for (int o = 128; o; o >>= 1) {
        if (tid < o) red[tid] += red[tid + o];
        __syncthreads();
    }
    const float L = m + logf(red[0]);
    float* orow = out + (size_t)b * Vc;
    for (int i = tid; i < Vc; i += 256) orow[i] = row[i] - L;
}

extern "C" void kernel_launch(void* const* d_in, const int* in_sizes, int n_in,
                              void* d_out, int out_size, void* d_ws,
                              size_t ws_size, hipStream_t stream) {
    const int*   x       = (const int*)  d_in[0];
    const float* h       = (const float*)d_in[1];   // [1,B,H]
    const float* enc     = (const float*)d_in[2];   // [T,B,H]
    const float* embW    = (const float*)d_in[3];   // [V,E]
    const float* attnW   = (const float*)d_in[4];   // [H,2H]
    const float* attnb   = (const float*)d_in[5];   // [H]
    const float* vvec    = (const float*)d_in[6];   // [H]
    const float* Wih     = (const float*)d_in[7];   // [3H, H+E]
    const float* Whh     = (const float*)d_in[8];   // [3H, H]
    const float* bih     = (const float*)d_in[9];
    const float* bhh     = (const float*)d_in[10];
    const float* fcW     = (const float*)d_in[11];  // [V, 2H]
    const float* fcb     = (const float*)d_in[12];

    float* out_logp = (float*)d_out;                 // [B,V]
    float* out_h    = out_logp + (size_t)Bc * Vc;    // [1,B,H]
    float* out_attn = out_h + (size_t)Bc * Hc;       // [B,1,T]

    float* ws       = (float*)d_ws;
    float* f_hwh    = ws;                            // B*H
    float* f_scores = f_hwh    + (size_t)Bc * Hc;    // B*T
    float* f_ctx    = f_scores + (size_t)Bc * Tc;    // B*H
    float* f_rnnin  = f_ctx    + (size_t)Bc * Hc;    // B*(E+H)
    float* f_gi     = f_rnnin  + (size_t)Bc * (Ec + Hc); // B*3H
    float* f_gh     = f_gi     + (size_t)Bc * 3 * Hc;    // B*3H
    float* f_hc     = f_gh     + (size_t)Bc * 3 * Hc;    // B*2H
    float* f_logits = f_hc     + (size_t)Bc * 2 * Hc;    // B*V

    // 1) hWh = h @ Wh^T + attn_b  (Wh = attn_W[:, :H])
    gemm16_wmma<<<dim3(Hc / 16, Bc / 16), 32, 0, stream>>>(
        h, Hc, attnW, 2 * Hc, attnb, f_hwh, Hc, Hc);

    // 2) fused energy-GEMM + relu + dot(v): 68.7 GFLOP bf16 WMMA
    attn_scores_wmma<<<dim3(BTc / 64), 128, 64 * Hc * 2, stream>>>(
        enc, attnW, f_hwh, vvec, f_scores);

    // 3) softmax over T -> attn weights (output tensor 3)
    softmax_T<<<dim3(Bc), 256, 0, stream>>>(f_scores, out_attn);

    // 4) context = attn_w @ enc
    context_k<<<dim3(Bc), 256, 0, stream>>>(enc, out_attn, f_ctx);

    // 5) rnn_in = [emb | context]
    build_rnnin<<<dim3(Bc * (Ec + Hc) / 256), 256, 0, stream>>>(
        x, embW, f_ctx, f_rnnin);

    // 6) GRU input/hidden GEMMs
    gemm16_wmma<<<dim3(3 * Hc / 16, Bc / 16), 32, 0, stream>>>(
        f_rnnin, Hc + Ec, Wih, Hc + Ec, bih, f_gi, 3 * Hc, Hc + Ec);
    gemm16_wmma<<<dim3(3 * Hc / 16, Bc / 16), 32, 0, stream>>>(
        h, Hc, Whh, Hc, bhh, f_gh, 3 * Hc, Hc);

    // 7) gates + h_new (output tensor 2) + hc = [h_new | context]
    gru_gates<<<dim3(Bc * Hc / 256), 256, 0, stream>>>(
        f_gi, f_gh, h, f_ctx, out_h, f_hc);

    // 8) logits = hc @ fc_W^T + fc_b (memory-bound on fc_W, 262 MB)
    gemm16_wmma<<<dim3(Vc / 16, Bc / 16), 32, 0, stream>>>(
        f_hc, 2 * Hc, fcW, 2 * Hc, fcb, f_logits, Vc, 2 * Hc);

    // 9) log_softmax -> output tensor 1
    logsoftmax_V<<<dim3(Bc), 256, 0, stream>>>(f_logits, out_logp);
}
